// LSTM_137438954428
// MI455X (gfx1250) — compile-verified, matrix operand order checked by reading, not run on hardware
//
#include <hip/hip_runtime.h>
#include <hip/hip_bf16.h>

// ---------------------------------------------------------------------------
// LSTM forward for MI455X (gfx1250, wave32, WMMA).
//   B=64, T=512, I=512, H=1024
// Phase 0: vectorized fp32->bf16 conversion of x, Wi, Wh (memory-bound).
// Phase 1: xg[t,b,g] = x[b,t,:] . Wi[g,:] + bias[g]
//          bf16 WMMA, f32 accum; each wave owns 32(M)x64(N): 8 WMMA/k-step.
// Phase 2: 512 sequential step kernels: gates = xg_t + h . Wh^T (bf16 WMMA),
//          fused sigmoid/tanh elementwise, h double-buffered in bf16.
// ---------------------------------------------------------------------------

typedef __bf16        v16bf __attribute__((ext_vector_type(16)));
typedef float         v8f   __attribute__((ext_vector_type(8)));
typedef unsigned int  v4u   __attribute__((ext_vector_type(4)));

#define B_   64
#define T_   512
#define I_   512
#define H_   1024
#define G4H  4096   // 4*H

union Frag { v16bf bf; v4u u[2]; };

__device__ __forceinline__ unsigned short f2bf(float f) {
    unsigned int u = __float_as_uint(f);
    unsigned int r = u + 0x7FFFu + ((u >> 16) & 1u);   // round-to-nearest-even
    return (unsigned short)(r >> 16);
}
__device__ __forceinline__ unsigned int pk2bf(float a, float b) {
    return (unsigned int)f2bf(a) | ((unsigned int)f2bf(b) << 16);
}
__device__ __forceinline__ float sig_(float x) {
    return 1.0f / (1.0f + __expf(-x));
}
__device__ __forceinline__ float tanh_(float x) {
    float ax = fabsf(x);
    float e  = __expf(-2.0f * ax);
    float r  = (1.0f - e) / (1.0f + e);
    return copysignf(r, x);
}

// ------------------------- phase 0: conversions ----------------------------
// 8 elements per thread: 2x b128 loads, 1x b128 store. n must be /8.
__global__ __launch_bounds__(256)
void cvt_f32_bf16_v8(const float* __restrict__ src,
                     unsigned short* __restrict__ dst, int n) {
    int i = (blockIdx.x * blockDim.x + threadIdx.x) * 8;
    if (i >= n) return;
    const float4 a = *(const float4*)(src + i);
    const float4 b = *(const float4*)(src + i + 4);
    v4u r;
    r[0] = pk2bf(a.x, a.y); r[1] = pk2bf(a.z, a.w);
    r[2] = pk2bf(b.x, b.y); r[3] = pk2bf(b.z, b.w);
    *(v4u*)(dst + i) = r;
}

__global__ __launch_bounds__(256)
void init_state(unsigned short* __restrict__ h0,
                float* __restrict__ c, int n) {          // n /8
    int i = (blockIdx.x * blockDim.x + threadIdx.x) * 8;
    if (i >= n) return;
    v4u z; z[0] = 0; z[1] = 0; z[2] = 0; z[3] = 0;
    *(v4u*)(h0 + i) = z;                                 // 8 bf16 zeros
    float4 zf = make_float4(0.f, 0.f, 0.f, 0.f);
    *(float4*)(c + i)     = zf;
    *(float4*)(c + i + 4) = zf;
}

// ------------------------- phase 1: input GEMM -----------------------------
// Grid: 8192 blocks x 256 threads (8 waves). Wave = 32(M) x 64(N) strip.
// M = b*T + t over x rows; output stored time-major [T, B, 4H].
__global__ __launch_bounds__(256)
void input_gemm(const unsigned short* __restrict__ xb,  // [B*T, I] bf16
                const unsigned short* __restrict__ wi,  // [4H, I]  bf16
                const float* __restrict__ bias,         // [4H]
                float* __restrict__ xg)                 // [T, B, 4H] f32
{
    const int lane  = threadIdx.x & 31;
    const int wid   = threadIdx.x >> 5;
    const int wg    = blockIdx.x * 8 + wid;     // 0 .. 65535
    const int mt    = wg >> 6;                  // 0 .. 1023 (32-row M tiles)
    const int ng    = wg & 63;                  // 0 .. 63   (64-col N strips)
    const int m0    = mt * 32;
    const int n0    = ng * 64;

    const int lrow  = lane & 15;
    const int kbase = (lane & 16) ? 8 : 0;      // 16-bit A/B layout K grouping

    const unsigned short* xrow0 = xb + (size_t)(m0 + lrow) * I_      + kbase;
    const unsigned short* xrow1 = xrow0 + 16 * I_;

    v8f acc[2][4];
    #pragma unroll
    for (int mi = 0; mi < 2; ++mi)
        #pragma unroll
        for (int j = 0; j < 4; ++j)
            #pragma unroll
            for (int q = 0; q < 8; ++q) acc[mi][j][q] = 0.0f;

    for (int k0 = 0; k0 < I_; k0 += 32) {
        Frag a0, a1;
        a0.u[0] = *(const v4u*)(xrow0 + k0);
        a0.u[1] = *(const v4u*)(xrow0 + k0 + 16);
        a1.u[0] = *(const v4u*)(xrow1 + k0);
        a1.u[1] = *(const v4u*)(xrow1 + k0 + 16);
        #pragma unroll
        for (int j = 0; j < 4; ++j) {
            const unsigned short* wrow =
                wi + (size_t)(n0 + j * 16 + lrow) * I_ + k0 + kbase;
            Frag bm;
            bm.u[0] = *(const v4u*)(wrow);
            bm.u[1] = *(const v4u*)(wrow + 16);
            acc[0][j] = __builtin_amdgcn_wmma_f32_16x16x32_bf16(
                false, a0.bf, false, bm.bf, (short)0, acc[0][j], false, false);
            acc[1][j] = __builtin_amdgcn_wmma_f32_16x16x32_bf16(
                false, a1.bf, false, bm.bf, (short)0, acc[1][j], false, false);
        }
    }

    // C/D layout: row = r + (lane>=16 ? 8:0), col = lane&15
    const int rofs = (lane & 16) ? 8 : 0;
    const int col  = lane & 15;
    const int bb   = m0 / T_;        // whole 32-row tile shares batch index
    const int tb   = m0 % T_;
    #pragma unroll
    for (int mi = 0; mi < 2; ++mi) {
        #pragma unroll
        for (int j = 0; j < 4; ++j) {
            const int n = n0 + j * 16 + col;
            const float bv = bias[n];
            #pragma unroll
            for (int r = 0; r < 8; ++r) {
                const int t = tb + mi * 16 + r + rofs;
                xg[((size_t)t * B_ + bb) * G4H + n] = acc[mi][j][r] + bv;
            }
        }
    }
}

// ------------------------- phase 2: recurrent step -------------------------
// Grid: 256 blocks x 32 threads (one wave per block -> spreads over WGPs).
// Wave computes one 16(batch) x 16(hidden) tile for all 4 gates, then fuses.
__global__ __launch_bounds__(32)
void lstm_step(const float* __restrict__ xg_t,            // [B,4H] slab for t
               const unsigned short* __restrict__ wh,     // [4H,H] bf16
               const unsigned short* __restrict__ h_in,   // [B,H] bf16
               unsigned short* __restrict__ h_out,        // [B,H] bf16
               float* __restrict__ cstate,                // [B,H] f32
               float* __restrict__ out_h,                 // [B,T,H] f32
               float* __restrict__ out_c,                 // [B,T,H] f32
               int t)
{
    const int lane  = threadIdx.x & 31;
    const int wg    = blockIdx.x;               // 0 .. 255
    const int b0    = (wg >> 6) * 16;           // 4 batch tiles
    const int h0    = (wg & 63) * 16;           // 64 hidden tiles
    const int lrow  = lane & 15;
    const int kbase = (lane & 16) ? 8 : 0;

    const unsigned short* hrow = h_in + (size_t)(b0 + lrow) * H_ + kbase;

    v8f acc[4];
    #pragma unroll
    for (int j = 0; j < 4; ++j)
        #pragma unroll
        for (int q = 0; q < 8; ++q) acc[j][q] = 0.0f;

    for (int k0 = 0; k0 < H_; k0 += 32) {
        Frag a;
        a.u[0] = *(const v4u*)(hrow + k0);
        a.u[1] = *(const v4u*)(hrow + k0 + 16);
        #pragma unroll
        for (int j = 0; j < 4; ++j) {
            const unsigned short* wrow =
                wh + (size_t)(j * H_ + h0 + lrow) * H_ + k0 + kbase;
            Frag bm;
            bm.u[0] = *(const v4u*)(wrow);
            bm.u[1] = *(const v4u*)(wrow + 16);
            acc[j] = __builtin_amdgcn_wmma_f32_16x16x32_bf16(
                false, a.bf, false, bm.bf, (short)0, acc[j], false, false);
        }
    }

    const int rofs = (lane & 16) ? 8 : 0;
    const int col  = lane & 15;
    const int h    = h0 + col;
    #pragma unroll
    for (int r = 0; r < 8; ++r) {
        const int b = b0 + r + rofs;
        const size_t gb = (size_t)b * G4H + h;
        const float i_ = sig_ (acc[0][r] + xg_t[gb + 0 * H_]);
        const float f_ = sig_ (acc[1][r] + xg_t[gb + 1 * H_]);
        const float g_ = tanh_(acc[2][r] + xg_t[gb + 2 * H_]);
        const float o_ = sig_ (acc[3][r] + xg_t[gb + 3 * H_]);
        const size_t s = (size_t)b * H_ + h;
        const float c_new = f_ * cstate[s] + i_ * g_;
        const float h_new = o_ * tanh_(c_new);
        cstate[s] = c_new;
        h_out[s]  = f2bf(h_new);
        const size_t o = ((size_t)b * T_ + t) * H_ + h;
        out_h[o] = h_new;
        out_c[o] = c_new;
    }
}

// ------------------------------ launcher -----------------------------------
extern "C" void kernel_launch(void* const* d_in, const int* in_sizes, int n_in,
                              void* d_out, int out_size, void* d_ws, size_t ws_size,
                              hipStream_t stream) {
    (void)in_sizes; (void)n_in; (void)out_size; (void)ws_size;

    const float* x    = (const float*)d_in[0];   // [B,T,I]
    const float* wi   = (const float*)d_in[1];   // [4H,I]
    const float* wh   = (const float*)d_in[2];   // [4H,H]
    const float* bias = (const float*)d_in[3];   // [4H]
    float* out = (float*)d_out;

    char* ws = (char*)d_ws;
    size_t off = 0;
    float* xg = (float*)(ws + off);                      off += (size_t)T_ * B_ * G4H * 4; // 512 MB
    unsigned short* xbf   = (unsigned short*)(ws + off); off += (size_t)B_ * T_ * I_ * 2;  // 32 MB
    unsigned short* wibf  = (unsigned short*)(ws + off); off += (size_t)G4H * I_ * 2;      // 4 MB
    unsigned short* whbf  = (unsigned short*)(ws + off); off += (size_t)G4H * H_ * 2;      // 8 MB
    unsigned short* hbuf0 = (unsigned short*)(ws + off); off += (size_t)B_ * H_ * 2;
    unsigned short* hbuf1 = (unsigned short*)(ws + off); off += (size_t)B_ * H_ * 2;
    float* cstate = (float*)(ws + off);                  off += (size_t)B_ * H_ * 4;

    // phase 0: conversions + state init (all sizes divisible by 8)
    const int NX = B_ * T_ * I_, NWI = G4H * I_, NWH = G4H * H_, NS = B_ * H_;
    cvt_f32_bf16_v8<<<NX  / (256 * 8), 256, 0, stream>>>(x,  xbf,  NX);
    cvt_f32_bf16_v8<<<NWI / (256 * 8), 256, 0, stream>>>(wi, wibf, NWI);
    cvt_f32_bf16_v8<<<NWH / (256 * 8), 256, 0, stream>>>(wh, whbf, NWH);
    init_state     <<<NS  / (256 * 8), 256, 0, stream>>>(hbuf0, cstate, NS);

    // phase 1: 1024 M-tiles(32) * 64 N-strips(64) = 65536 waves / 8 per block
    input_gemm<<<8192, 256, 0, stream>>>(xbf, wibf, bias, xg);

    // phase 2: sequential recurrence, h double-buffered
    float* out_h = out;
    float* out_c = out + (size_t)B_ * T_ * H_;
    for (int t = 0; t < T_; ++t) {
        const unsigned short* hin = (t & 1) ? hbuf1 : hbuf0;
        unsigned short*      hout = (t & 1) ? hbuf0 : hbuf1;
        lstm_step<<<256, 32, 0, stream>>>(xg + (size_t)t * B_ * G4H,
                                          whbf, hin, hout, cstate,
                                          out_h, out_c, t);
    }
}
